// GRU_Predictor_69801808494929
// MI455X (gfx1250) — matrix operand order repaired and checked by reading, not compiled, not run on hardware
//
#include <hip/hip_runtime.h>
#include <hip/hip_bf16.h>
#include <stdint.h>

// GRU: B=512, T=2048, I=8, H=128, O=96
#define BB 512
#define TT 2048
#define II 8
#define HH 128
#define OO 96

#define AST 168   // bf16 elements per row of augmented-h LDS buffer (160 used + 8 pad -> 336B stride, conflict-free)
#define PF  8     // x prefetch ring depth (async-to-LDS)

typedef __bf16        bf16x16 __attribute__((ext_vector_type(16)));
typedef float         f32x8   __attribute__((ext_vector_type(8)));
typedef unsigned int  u32x8   __attribute__((ext_vector_type(8)));

static __device__ __forceinline__ unsigned pkbf(float lo, float hi) {
  union { float f; unsigned u; } a, b; a.f = lo; b.f = hi;
  unsigned ul = a.u + 0x7FFFu + ((a.u >> 16) & 1u);   // RNE
  unsigned uh = b.u + 0x7FFFu + ((b.u >> 16) & 1u);
  return (ul >> 16) | (uh & 0xFFFF0000u);
}
static __device__ __forceinline__ unsigned short bf1(float x) {
  union { float f; unsigned u; } a; a.f = x;
  return (unsigned short)((a.u + 0x7FFFu + ((a.u >> 16) & 1u)) >> 16);
}

// Fast gate nonlinearities: single-instruction v_rcp_f32 / v_tanh_f32.
static __device__ __forceinline__ float frcp(float x) { return __builtin_amdgcn_rcpf(x); }

#if __has_builtin(__builtin_amdgcn_tanhf)
static __device__ __forceinline__ float tanh_hw(float x) { return __builtin_amdgcn_tanhf(x); }
static __device__ __forceinline__ float sigm(float x)    { return fmaf(0.5f, tanh_hw(0.5f * x), 0.5f); }
#else
static __device__ __forceinline__ float tanh_hw(float x) { return 1.f - 2.f * frcp(1.f + __expf(2.f * x)); }
static __device__ __forceinline__ float sigm(float x)    { return frcp(1.f + __expf(-x)); }
#endif

static __device__ __forceinline__ f32x8 wmma_bf16(u32x8 a, u32x8 b, f32x8 c) {
  return __builtin_amdgcn_wmma_f32_16x16x32_bf16(
      false, __builtin_bit_cast(bf16x16, a),
      false, __builtin_bit_cast(bf16x16, b),
      (short)0, c, false, false);
}

// Pack 16 consecutive fp32 -> bf16 B-fragment half (K ascending, 2 per VGPR).
static __device__ __forceinline__ u32x8 pack16(const float* p) {
  float4 f0 = ((const float4*)p)[0], f1 = ((const float4*)p)[1],
         f2 = ((const float4*)p)[2], f3 = ((const float4*)p)[3];
  u32x8 b;
  b[0]=pkbf(f0.x,f0.y); b[1]=pkbf(f0.z,f0.w);
  b[2]=pkbf(f1.x,f1.y); b[3]=pkbf(f1.z,f1.w);
  b[4]=pkbf(f2.x,f2.y); b[5]=pkbf(f2.z,f2.w);
  b[6]=pkbf(f3.x,f3.y); b[7]=pkbf(f3.z,f3.w);
  return b;
}

// A fragment (16x32 bf16) from LDS row base: v0..7 = K kb..kb+15 (contiguous).
static __device__ __forceinline__ u32x8 loadA(const unsigned short* rbase, int kf) {
  const uint4* pa = (const uint4*)(rbase + kf * 32);
  uint4 lo = pa[0], hi = pa[1];
  u32x8 a;
  a[0]=lo.x; a[1]=lo.y; a[2]=lo.z; a[3]=lo.w;
  a[4]=hi.x; a[5]=hi.y; a[6]=hi.z; a[7]=hi.w;
  return a;
}

// gx_n for one step: per-lane dot of an x row (LDS, half-wave-uniform -> broadcast) with w_ih_n.
static __device__ __forceinline__ void gxn_dot(float* dst, const float* xslot,
                                               const float* wn, float bn, int hi16) {
#pragma unroll
  for (int v = 0; v < 8; ++v) {
    const float* xr = xslot + (v + 8*hi16) * II;
    float4 xa = *(const float4*)xr;
    float s = fmaf(xa.x, wn[0], bn);
    s = fmaf(xa.y, wn[1], s);
    s = fmaf(xa.z, wn[2], s);
    s = fmaf(xa.w, wn[3], s);
    float4 xb = *(const float4*)(xr + 4);
    s = fmaf(xb.x, wn[4], s);
    s = fmaf(xb.y, wn[5], s);
    s = fmaf(xb.z, wn[6], s);
    s = fmaf(xb.w, wn[7], s);
    dst[v] = s;
  }
}

__global__ __launch_bounds__(256) void gru_fused_kernel(
    const float* __restrict__ x,    const float* __restrict__ w_ih,
    const float* __restrict__ w_hh, const float* __restrict__ b_ih,
    const float* __restrict__ b_hh, const float* __restrict__ fc_w,
    const float* __restrict__ fc_b, float* __restrict__ out)
{
  __shared__ unsigned short sbuf[2][16 * AST];   // ping-pong augmented A = [h | x_t | 0-pad], bf16
  __shared__ float          xs[PF][16][II];      // fp32 x staging ring (filled by async DMA)

  const int tid  = threadIdx.x;
  const int lane = tid & 31;
  const int wv   = tid >> 5;        // 0..7
  const int l15  = lane & 15;
  const int hi16 = lane >> 4;       // 0 | 1
  const int b0   = blockIdx.x * 16; // batch tile base

  // ---------------- prologue ----------------
  { // zero h region of buf0 (h0 == 0): 16 rows x 128 cols
    int row = tid >> 4, c8 = (tid & 15) * 8;
    *(uint4*)&sbuf[0][row * AST + c8] = make_uint4(0u, 0u, 0u, 0u);
  }
  if (tid < 16) { // tail cols [128..AST): zero both buffers (x written per-step; rest stays 0)
    for (int c = 128; c < AST; ++c) { sbuf[0][tid*AST + c] = 0; sbuf[1][tid*AST + c] = 0; }
  }

  // kick off x prefetch for steps 0..PF-2, then build x-tail for t=0 into buf0
  if (wv == 0 && lane < 16) {
    for (int f = 0; f < PF - 1; ++f) {
      unsigned la = (unsigned)(uintptr_t)&xs[f & (PF - 1)][lane][0];
      unsigned vo = (unsigned)(((b0 + lane) * TT + f) * II * 4);
      asm volatile("global_load_async_to_lds_b128 %0, %1, %2 offset:0"  :: "v"(la), "v"(vo), "s"(x) : "memory");
      asm volatile("global_load_async_to_lds_b128 %0, %1, %2 offset:16" :: "v"(la), "v"(vo), "s"(x) : "memory");
    }
    asm volatile("s_wait_asynccnt 0" ::: "memory");
    const float* xr = &xs[0][lane][0];
    *(uint4*)&sbuf[0][lane * AST + 128] =
        make_uint4(pkbf(xr[0],xr[1]), pkbf(xr[2],xr[3]), pkbf(xr[4],xr[5]), pkbf(xr[6],xr[7]));
  }

  // ---- register-resident augmented weights: B[k][n], n = gate column, K = 160 (h|x|pad) ----
  // This wave owns cols [16*wv, 16*wv+16) of each gate (r, z, n). Biases go into the C-init.
  const int gc = 16*wv + l15;
  const int gr = gc, gz = HH + gc, gn = 2*HH + gc;

  u32x8 BfR[5], BfZ[5], BfN[4];
#pragma unroll
  for (int kf = 0; kf < 4; ++kf) {
    BfR[kf] = pack16(w_hh + (size_t)gr * HH + kf*32 + hi16*16);
    BfZ[kf] = pack16(w_hh + (size_t)gz * HH + kf*32 + hi16*16);
    BfN[kf] = pack16(w_hh + (size_t)gn * HH + kf*32 + hi16*16);
  }
  { // kf=4 tails: rows 128..135 = w_ih.T for r,z (x contribution); n-gate handled via VALU dot
    u32x8 tR = {}, tZ = {};
    if (hi16 == 0) {
      const float* wiR = w_ih + (size_t)gr * II;
      tR[0]=pkbf(wiR[0],wiR[1]); tR[1]=pkbf(wiR[2],wiR[3]);
      tR[2]=pkbf(wiR[4],wiR[5]); tR[3]=pkbf(wiR[6],wiR[7]);
      const float* wiZ = w_ih + (size_t)gz * II;
      tZ[0]=pkbf(wiZ[0],wiZ[1]); tZ[1]=pkbf(wiZ[2],wiZ[3]);
      tZ[2]=pkbf(wiZ[4],wiZ[5]); tZ[3]=pkbf(wiZ[6],wiZ[7]);
    }
    BfR[4] = tR; BfZ[4] = tZ;
  }

  const float cbr = b_ih[gr] + b_hh[gr];   // r-gate bias -> C-init
  const float cbz = b_ih[gz] + b_hh[gz];   // z-gate bias -> C-init
  const float cbn = b_hh[gn];              // n-gate recurrent bias -> C-init

  // per-lane w_ih row for the n-gate column this lane owns (separate gx_n dot)
  float wn[8];
#pragma unroll
  for (int i = 0; i < 8; ++i) wn[i] = w_ih[gn * II + i];
  const float bn = b_ih[gn];

  float h[8];
#pragma unroll
  for (int v = 0; v < 8; ++v) h[v] = 0.f;

  __syncthreads();

  // gx_n software pipeline: gxnC holds the value for the current step.
  float gxnC[8], gxnN[8];
  gxn_dot(gxnC, &xs[0][0][0], wn, bn, hi16);
#pragma unroll
  for (int v = 0; v < 8; ++v) gxnN[v] = 0.f;

  // ---------------- recurrent scan ----------------
#pragma unroll 1
  for (int t = 0; t < TT; ++t) {
    const int rb = t & 1, wb = rb ^ 1;

    // async prefetch x for step t+PF-1 into ring slot (overwrites slot consumed 2 barriers ago)
    if (wv == 0 && lane < 16) {
      int f = t + PF - 1;
      if (f < TT) {
        unsigned la = (unsigned)(uintptr_t)&xs[f & (PF - 1)][lane][0];
        unsigned vo = (unsigned)(((b0 + lane) * TT + f) * II * 4);
        asm volatile("global_load_async_to_lds_b128 %0, %1, %2 offset:0"  :: "v"(la), "v"(vo), "s"(x) : "memory");
        asm volatile("global_load_async_to_lds_b128 %0, %1, %2 offset:16" :: "v"(la), "v"(vo), "s"(x) : "memory");
      }
    }

    // gx_n for step t+1 (depends only on x; overlaps this step's WMMA work).
    // Slot t+1 is guaranteed landed: previous iteration waited asynccnt<=10 (completes t+1, t+2).
    if (t + 1 < TT) gxn_dot(gxnN, &xs[(t + 1) & (PF - 1)][0][0], wn, bn, hi16);

    // 14 WMMAs, A fragment loaded per-K-slice to keep liveness low.
    // r,z: gh + gx + bias (bias via C-init, x via augmented K); n: gh + b_hh only.
    const unsigned short* rbase = &sbuf[rb][l15 * AST + hi16 * 16];
    f32x8 aR, aZ, aN;
#pragma unroll
    for (int i = 0; i < 8; ++i) { aR[i] = cbr; aZ[i] = cbz; aN[i] = cbn; }
#pragma unroll
    for (int kf = 0; kf < 4; ++kf) {
      u32x8 a = loadA(rbase, kf);
      aR = wmma_bf16(a, BfR[kf], aR);
      aZ = wmma_bf16(a, BfZ[kf], aZ);
      aN = wmma_bf16(a, BfN[kf], aN);
    }
    {
      u32x8 a4 = loadA(rbase, 4);            // [x_t | zeros] tail
      aR = wmma_bf16(a4, BfR[4], aR);
      aZ = wmma_bf16(a4, BfZ[4], aZ);
    }

    // gate nonlinearities + state update; publish bf16 h slice to write buffer
#pragma unroll
    for (int v = 0; v < 8; ++v) {
      float r = sigm(aR[v]);
      float z = sigm(aZ[v]);
      float n = tanh_hw(gxnC[v] + r * aN[v]);
      h[v] = (1.f - z) * n + z * h[v];
      sbuf[wb][(v + 8*hi16) * AST + wv*16 + l15] = bf1(h[v]);
    }

    // rotate the gx_n pipeline
#pragma unroll
    for (int v = 0; v < 8; ++v) gxnC[v] = gxnN[v];

    // wave0: guarantee slots t+1 AND t+2 have landed (so next iteration's gxn_dot and
    // tail-build may read slot t+2 after the barrier), build x-tail for t+1.
    if (wv == 0) {
      asm volatile("s_wait_asynccnt 10" ::: "memory");
      if (lane < 16 && t + 1 < TT) {
        const float* xr = &xs[(t + 1) & (PF - 1)][lane][0];
        *(uint4*)&sbuf[wb][lane * AST + 128] =
            make_uint4(pkbf(xr[0],xr[1]), pkbf(xr[2],xr[3]), pkbf(xr[4],xr[5]), pkbf(xr[6],xr[7]));
      }
    }
    __syncthreads();
  }

  // ---------------- final FC: out = h_T @ fc_w.T + fc_b ----------------
  if (wv < 6) {                               // 6 tiles cover O=96
    const int o = wv * 16 + l15;
    const unsigned short* rbase = &sbuf[TT & 1][l15 * AST + hi16 * 16];
    f32x8 acc = {};
#pragma unroll
    for (int kf = 0; kf < 4; ++kf) {
      u32x8 bo = pack16(fc_w + (size_t)o * HH + kf*32 + hi16*16);
      acc = wmma_bf16(loadA(rbase, kf), bo, acc);
    }
    const float bo = fc_b[o];
#pragma unroll
    for (int v = 0; v < 8; ++v)
      out[(size_t)(b0 + v + 8*hi16) * OO + o] = acc[v] + bo;
  }
}

extern "C" void kernel_launch(void* const* d_in, const int* in_sizes, int n_in,
                              void* d_out, int out_size, void* d_ws, size_t ws_size,
                              hipStream_t stream) {
  (void)in_sizes; (void)n_in; (void)out_size; (void)d_ws; (void)ws_size;
  const float* x    = (const float*)d_in[0];
  const float* w_ih = (const float*)d_in[1];
  const float* w_hh = (const float*)d_in[2];
  const float* b_ih = (const float*)d_in[3];
  const float* b_hh = (const float*)d_in[4];
  const float* fc_w = (const float*)d_in[5];
  const float* fc_b = (const float*)d_in[6];
  gru_fused_kernel<<<dim3(BB / 16), dim3(256), 0, stream>>>(
      x, w_ih, w_hh, b_ih, b_hh, fc_w, fc_b, (float*)d_out);
}
